// LSTMPredictor_85461259255839
// MI455X (gfx1250) — matrix operand order repaired and checked by reading, not compiled, not run on hardware
//
#include <hip/hip_runtime.h>
#include <hip/hip_bf16.h>

// ---------------------------------------------------------------------------
// Fused 3-layer LSTM (H=50, P=17) for MI455X / gfx1250.
// One workgroup (256 threads = 8 wave32) owns BM=16 batch rows and runs the
// full T+future recurrence with weights + state resident in LDS (~184 KB of
// the 320 KB/WGP pool).  All GEMMs are v_wmma_f32_16x16x32_f16; activations
// use raw v_exp_f32 / v_rcp_f32; wave index is scalarized via readfirstlane
// so tile loops are SALU-controlled (EXEC stays all-ones around WMMA).
// ---------------------------------------------------------------------------

#define THREADS 256
#define NWAVES  8
#define BM      16    // batch rows per workgroup
#define P       17    // prediction chunk
#define H       50    // hidden
#define NG      200   // 4*H gate width
#define NP      208   // padded gate width (13 tiles of 16)
#define NTILES  13
#define KX      32    // padded K for x-input (17 -> 32)
#define KH      64    // padded K for hidden (50 -> 64)

typedef __attribute__((ext_vector_type(16))) _Float16 v16h;
typedef __attribute__((ext_vector_type(8)))  float    v8f;
typedef __attribute__((ext_vector_type(2)))  _Float16 half2v;

#define WMMA_F16(a, b, c) \
  __builtin_amdgcn_wmma_f32_16x16x32_f16(false, (a), false, (b), (short)0, (c), false, false)

// LDS working set (per workgroup).  Weights stored transposed [n][k] (f16,
// zero padded) so B-fragment loads are two contiguous 16B runs per lane.
struct Smem {
  _Float16 Wih1[NP * KX];
  _Float16 Whh1[NP * KH];
  _Float16 Wih2[NP * KH];
  _Float16 Whh2[NP * KH];
  _Float16 Wih3[NP * KH];
  _Float16 Whh3[NP * KH];
  _Float16 Wlin[32 * KH];
  float bsum1[NP];
  float bsum2[NP];
  float bsum3[NP];
  float blin[32];
  _Float16 xbuf[BM * KX];           // layer-1 input (x chunk, or fed-back out)
  _Float16 h1[BM * KH];
  _Float16 h2[BM * KH];
  _Float16 h3[BM * KH];
  float c1[BM * KH];
  float c2[BM * KH];
  float c3[BM * KH];
  float gates[BM * NP];             // WMMA accumulator staging
  float outb[BM * 32];              // linear head staging
};

// Fast activations on the v_exp_f32 / v_rcp_f32 hardware paths.
// rcp(inf)=0 gives correct saturation at the tails.
__device__ __forceinline__ float fast_rcp(float xv) { return __builtin_amdgcn_rcpf(xv); }
__device__ __forceinline__ float sigm(float xv) {
  return fast_rcp(1.0f + __expf(-xv));
}
__device__ __forceinline__ float fast_tanh(float xv) {
  return 1.0f - 2.0f * fast_rcp(1.0f + __expf(2.0f * xv));
}

// A fragment: 16x32 f16, rows striped over lanes 0..15 (==16..31),
// VGPR v holds K pair; lanes>=16 shifted by K+8 (ISA 7.12.2 layout).
__device__ __forceinline__ v16h load_A(const _Float16* __restrict__ base, int stride, int lane) {
  const _Float16* row = base + (lane & 15) * stride;
  const int koff = (lane >= 16) ? 8 : 0;
  v16h a;
#pragma unroll
  for (int vv = 0; vv < 8; ++vv) {
    int k = ((vv < 4) ? (2 * vv) : (16 + 2 * (vv - 4))) + koff;
    half2v pr = *(const half2v*)(row + k);
    a[2 * vv]     = pr[0];
    a[2 * vv + 1] = pr[1];
  }
  return a;
}

// B fragment: 32x16 f16 (KxN).  Weights stored [n][k]; same K order as A:
// lanes 0..15 hold K=0..7,16..23, lanes 16..31 hold K=8..15,24..31.
__device__ __forceinline__ v16h load_B(const _Float16* __restrict__ base, int stride, int lane) {
  const _Float16* col = base + (lane & 15) * stride + ((lane >= 16) ? 8 : 0);
  v16h b;
#pragma unroll
  for (int vv = 0; vv < 8; ++vv) {
    int k = (vv < 4) ? (2 * vv) : (16 + 2 * (vv - 4));
    half2v pr = *(const half2v*)(col + k);
    b[2 * vv]     = pr[0];
    b[2 * vv + 1] = pr[1];
  }
  return b;
}

// C/D fragment scatter to LDS: VGPR r -> row r (+8 for lanes>=16), col = lane&15.
// Per ds_store instruction only n varies across lanes -> conflict-free.
__device__ __forceinline__ void store_D(float* __restrict__ buf, int stride, int ntile,
                                        v8f d, int lane) {
  float* p = buf + ((lane >= 16) ? 8 : 0) * stride + ntile * 16 + (lane & 15);
#pragma unroll
  for (int r = 0; r < 8; ++r) p[r * stride] = d[r];
}

// gates[16x208] = [hin | hself] x [Wih ; Whh]   (hidden->gates layers)
__device__ __forceinline__ void gates_mm_h(const _Float16* hin, const _Float16* hself,
                                           const _Float16* Wih, const _Float16* Whh,
                                           float* gates, int wave, int lane) {
  v16h a0 = load_A(hin, KH, lane);
  v16h a1 = load_A(hin + 32, KH, lane);
  v16h a2 = load_A(hself, KH, lane);
  v16h a3 = load_A(hself + 32, KH, lane);
  for (int nt = wave; nt < NTILES; nt += NWAVES) {   // nt is SGPR-uniform
    const _Float16* wi = Wih + nt * 16 * KH;
    const _Float16* wh = Whh + nt * 16 * KH;
    v8f acc = {};
    acc = WMMA_F16(a0, load_B(wi,      KH, lane), acc);
    acc = WMMA_F16(a1, load_B(wi + 32, KH, lane), acc);
    acc = WMMA_F16(a2, load_B(wh,      KH, lane), acc);
    acc = WMMA_F16(a3, load_B(wh + 32, KH, lane), acc);
    store_D(gates, NP, nt, acc, lane);
  }
}

// gates[16x208] = [x | h1] x [Wih1 ; Whh1]   (layer 1: K=17->32 for x)
__device__ __forceinline__ void gates_mm_x(const _Float16* xb, const _Float16* hself,
                                           const _Float16* Wih, const _Float16* Whh,
                                           float* gates, int wave, int lane) {
  v16h a0 = load_A(xb, KX, lane);
  v16h a2 = load_A(hself, KH, lane);
  v16h a3 = load_A(hself + 32, KH, lane);
  for (int nt = wave; nt < NTILES; nt += NWAVES) {   // nt is SGPR-uniform
    const _Float16* wh = Whh + nt * 16 * KH;
    v8f acc = {};
    acc = WMMA_F16(a0, load_B(Wih + nt * 16 * KX, KX, lane), acc);
    acc = WMMA_F16(a2, load_B(wh,      KH, lane), acc);
    acc = WMMA_F16(a3, load_B(wh + 32, KH, lane), acc);
    store_D(gates, NP, nt, acc, lane);
  }
}

// c,h update over the PADDED [BM][KH] domain: exactly 4 unrolled iterations,
// row = idx>>6, u = idx&63 (pure shift/mask), c/h indexed directly by idx.
// torch gate order i,f,g,o at offsets 0,H,2H,3H within a gates row.
__device__ __forceinline__ void lstm_pointwise(const float* gates, const float* bsum,
                                               float* c, _Float16* h, int tid) {
#pragma unroll
  for (int i = 0; i < (BM * KH) / THREADS; ++i) {
    int idx = tid + i * THREADS;
    int row = idx >> 6;
    int u   = idx & (KH - 1);
    if (u < H) {
      const float* g = gates + row * NP;
      float gi = g[u]         + bsum[u];
      float gf = g[u + H]     + bsum[u + H];
      float gg = g[u + 2 * H] + bsum[u + 2 * H];
      float go = g[u + 3 * H] + bsum[u + 3 * H];
      float cv = c[idx];
      float cn = sigm(gf) * cv + sigm(gi) * fast_tanh(gg);
      float hn = sigm(go) * fast_tanh(cn);
      c[idx] = cn;
      h[idx] = (_Float16)hn;
    }
  }
}

// f32 global weight [Nv][Kv] -> f16 LDS transposed-padded [Npad][Kpad]
__device__ __forceinline__ void stage_w(const float* __restrict__ src, _Float16* __restrict__ dst,
                                        int Nv, int Kv, int Npad, int Kpad, int tid) {
  for (int idx = tid; idx < Npad * Kpad; idx += THREADS) {
    int n = idx / Kpad, k = idx - n * Kpad;
    float v = (n < Nv && k < Kv) ? src[n * Kv + k] : 0.0f;
    dst[idx] = (_Float16)v;
  }
}

__global__ __launch_bounds__(THREADS)
void lstm_fused(const float* __restrict__ x,
                const float* __restrict__ Wih1, const float* __restrict__ Whh1,
                const float* __restrict__ bih1, const float* __restrict__ bhh1,
                const float* __restrict__ Wih2, const float* __restrict__ Whh2,
                const float* __restrict__ bih2, const float* __restrict__ bhh2,
                const float* __restrict__ Wih3, const float* __restrict__ Whh3,
                const float* __restrict__ bih3, const float* __restrict__ bhh3,
                const float* __restrict__ Wlin, const float* __restrict__ blinG,
                const int* __restrict__ future,
                float* __restrict__ out, int T) {
  extern __shared__ char smem_raw[];
  Smem& S = *reinterpret_cast<Smem*>(smem_raw);
  const int tid  = threadIdx.x;
  const int lane = tid & 31;
  // Scalarize the wave index: uniform within a wave by construction, and
  // readfirstlane lets the compiler keep tile loops in SALU control flow
  // (no exec-mask vector-loop pattern around v_wmma).
  const int wave = __builtin_amdgcn_readfirstlane(tid >> 5);
  const int b0   = blockIdx.x * BM;

  // ---- one-time LDS staging -------------------------------------------------
  stage_w(Wih1, S.Wih1, NG, P, NP, KX, tid);
  stage_w(Whh1, S.Whh1, NG, H, NP, KH, tid);
  stage_w(Wih2, S.Wih2, NG, H, NP, KH, tid);
  stage_w(Whh2, S.Whh2, NG, H, NP, KH, tid);
  stage_w(Wih3, S.Wih3, NG, H, NP, KH, tid);
  stage_w(Whh3, S.Whh3, NG, H, NP, KH, tid);
  stage_w(Wlin, S.Wlin, P,  H, 32, KH, tid);
  for (int i = tid; i < NP; i += THREADS) {
    S.bsum1[i] = (i < NG) ? bih1[i] + bhh1[i] : 0.0f;
    S.bsum2[i] = (i < NG) ? bih2[i] + bhh2[i] : 0.0f;
    S.bsum3[i] = (i < NG) ? bih3[i] + bhh3[i] : 0.0f;
  }
  if (tid < 32) S.blin[tid] = (tid < P) ? blinG[tid] : 0.0f;
  for (int i = tid; i < BM * KH; i += THREADS) {
    S.h1[i] = (_Float16)0.0f; S.h2[i] = (_Float16)0.0f; S.h3[i] = (_Float16)0.0f;
    S.c1[i] = 0.0f;           S.c2[i] = 0.0f;           S.c3[i] = 0.0f;
  }
  for (int i = tid; i < BM * KX; i += THREADS) S.xbuf[i] = (_Float16)0.0f;

  const int total = T + future[0] / P;      // 512 + 32 closed-loop steps
  const size_t xrow = (size_t)T * P;
  const size_t orow = (size_t)total * P;

  __syncthreads();

  // ---- sequential recurrence, fully LDS-resident ---------------------------
  for (int t = 0; t < total; ++t) {
    if (t < T) {                            // teacher-forced input; else xbuf
      for (int idx = tid; idx < BM * KX; idx += THREADS) {  // holds fed-back out
        int row = idx >> 5, k = idx & 31;
        float v = (k < P) ? x[(size_t)(b0 + row) * xrow + (size_t)t * P + k] : 0.0f;
        S.xbuf[idx] = (_Float16)v;
      }
    }
    __syncthreads();
    gates_mm_x(S.xbuf, S.h1, S.Wih1, S.Whh1, S.gates, wave, lane);
    __syncthreads();
    lstm_pointwise(S.gates, S.bsum1, S.c1, S.h1, tid);
    __syncthreads();
    gates_mm_h(S.h1, S.h2, S.Wih2, S.Whh2, S.gates, wave, lane);
    __syncthreads();
    lstm_pointwise(S.gates, S.bsum2, S.c2, S.h2, tid);
    __syncthreads();
    gates_mm_h(S.h2, S.h3, S.Wih3, S.Whh3, S.gates, wave, lane);
    __syncthreads();
    lstm_pointwise(S.gates, S.bsum3, S.c3, S.h3, tid);
    __syncthreads();

    // linear head: out[16x17] = h3[16x50] @ Wlin^T   (2 N-tiles, waves 0/1)
    if (wave < 2) {                          // scalar branch
      v16h a0 = load_A(S.h3, KH, lane);
      v16h a1 = load_A(S.h3 + 32, KH, lane);
      const _Float16* w = S.Wlin + wave * 16 * KH;
      v8f acc = {};
      acc = WMMA_F16(a0, load_B(w,      KH, lane), acc);
      acc = WMMA_F16(a1, load_B(w + 32, KH, lane), acc);
      store_D(S.outb, 32, wave, acc, lane);
    }
    __syncthreads();

    // emit output and feed it back as next x (used once t >= T)
    for (int idx = tid; idx < BM * 32; idx += THREADS) {
      int row = idx >> 5, k = idx & 31;
      float v = (k < P) ? (S.outb[idx] + S.blin[k]) : 0.0f;
      if (k < P) out[(size_t)(b0 + row) * orow + (size_t)t * P + k] = v;
      S.xbuf[idx] = (_Float16)v;
    }
    __syncthreads();
  }
}

extern "C" void kernel_launch(void* const* d_in, const int* in_sizes, int n_in,
                              void* d_out, int out_size, void* d_ws, size_t ws_size,
                              hipStream_t stream) {
  const float* x    = (const float*)d_in[0];
  const float* Wih1 = (const float*)d_in[1];
  const float* Whh1 = (const float*)d_in[2];
  const float* bih1 = (const float*)d_in[3];
  const float* bhh1 = (const float*)d_in[4];
  const float* Wih2 = (const float*)d_in[5];
  const float* Whh2 = (const float*)d_in[6];
  const float* bih2 = (const float*)d_in[7];
  const float* bhh2 = (const float*)d_in[8];
  const float* Wih3 = (const float*)d_in[9];
  const float* Whh3 = (const float*)d_in[10];
  const float* bih3 = (const float*)d_in[11];
  const float* bhh3 = (const float*)d_in[12];
  const float* Wlin = (const float*)d_in[13];
  const float* blin = (const float*)d_in[14];
  const int*   fut  = (const int*)d_in[15];

  const int B = 2048;
  const int T = in_sizes[0] / (B * P);      // 512

  dim3 grid(B / BM), block(THREADS);
  lstm_fused<<<grid, block, sizeof(Smem), stream>>>(
      x, Wih1, Whh1, bih1, bhh1, Wih2, Whh2, bih2, bhh2,
      Wih3, Whh3, bih3, bhh3, Wlin, blin, fut, (float*)d_out, T);
}
